// XbarConv_69123203662535
// MI455X (gfx1250) — compile-verified, matrix operand order checked by reading, not run on hardware
//
#include <hip/hip_runtime.h>
#include <hip/hip_bf16.h>

typedef __attribute__((ext_vector_type(16))) _Float16 v16h;
typedef __attribute__((ext_vector_type(8)))  float    v8f;

#define BATCH 16
#define IC 8
#define H 256
#define W 256
#define OC 16
#define OH 254
#define OW 254

#define TROWS 16
#define TCOLS 64
#define IN_ROWS 18              // TROWS + 2
#define IN_COLS 66              // TCOLS + 2
// LDS layout: channel-last, one dword = 4 adjacent channels of one (r,col):
//   xq32[(r*IN_COLS + col)*2 + hf]  holds channels 4*hf .. 4*hf+3 (bytes 0..3)
#define LDS_U32 (IN_ROWS * IN_COLS * 2)   // 2376 dwords = 9504 B

// One workgroup = 8 waves; each wave owns 8 groups of 16 output pixels
// (16 oc x 16 px per WMMA tile). K-dim = im2col, K = q*8 + i with tap
// q = kh*3+kw (0..8 real, 9..11 zero-pad -> K = 96 = 3 x k32 WMMA steps)
// and channel(i) = perm[i], perm = {0,2,1,3,4,6,5,7}, chosen so a single
// channel-last LDS dword feeds two f16 fragment halves per shift/mask/mul.
__global__ __launch_bounds__(256)
void xbarconv_wmma_kernel(const float* __restrict__ x,
                          const float* __restrict__ wgt,
                          float* __restrict__ out)
{
    __shared__ unsigned int xq32[LDS_U32];

    const int tid    = threadIdx.x;
    const int lane   = tid & 31;
    const int waveId = tid >> 5;
    const int laneHi = (lane >> 4) & 1;   // 0: lanes 0-15, 1: lanes 16-31
    const int mn     = lane & 15;         // A/D row (M) ; B/D col (N)

    const int colTile = blockIdx.x;       // 0..3
    const int rowTile = blockIdx.y;       // 0..15
    const int b       = blockIdx.z;       // 0..15

    const int oh0 = rowTile * TROWS;
    const int ow0 = colTile * TCOLS;

    // ---- per-lane tap offsets (dword units in xq32) -------------------
    // half h -> K(h) = 32*s + (h<8?0:16) + 8*laneHi + (h&7)
    //   tap q = 2*j + laneHi, j = 2*s + (h<8?0:1); q>8 = zero-pad taps:
    //   weights forced to 0, so patch bytes may be garbage (clamped addr).
    int rc2[6];
#pragma unroll
    for (int j = 0; j < 6; ++j) {
        int q  = 2 * j + laneHi;
        int qc = q > 8 ? 8 : q;
        rc2[j] = ((qc / 3) * IN_COLS + (qc % 3)) * 2;
    }

    // ---- A fragments: f16 weights pre-scaled by 1/16 (quant divisor) --
    const int perm[8] = {0, 2, 1, 3, 4, 6, 5, 7};
    union { v16h v; _Float16 f[16]; } A[3];
#pragma unroll
    for (int s = 0; s < 3; ++s) {
#pragma unroll
        for (int h = 0; h < 16; ++h) {
            const int c = perm[h & 7];
            const int q = 4 * s + (h < 8 ? 0 : 2) + laneHi;
            float wv = (q < 9) ? wgt[(mn * IC + c) * 9 + q] * 0.0625f : 0.0f;
            A[s].f[h] = (_Float16)wv;
        }
    }

    // ---- Phase 1: clamp + int8-cast, 4 channels packed per dword ------
    for (int idx = tid; idx < IN_ROWS * IN_COLS * 2; idx += 256) {
        const int hf  = idx & 1;          // channel group: 4*hf .. 4*hf+3
        const int rc  = idx >> 1;
        const int r   = rc / IN_COLS;
        const int col = rc - r * IN_COLS;
        int gr = oh0 + r;   gr = gr > (H - 1) ? (H - 1) : gr;
        int gc = ow0 + col; gc = gc > (W - 1) ? (W - 1) : gc;
        const float* px = x + ((b * IC + hf * 4) * H + gr) * W + gc;
        unsigned int packed = 0;
#pragma unroll
        for (int c = 0; c < 4; ++c) {
            float v = __builtin_amdgcn_fmed3f(px[c * H * W], -128.0f, 127.0f);
            packed |= ((unsigned int)(int)v & 0xFFu) << (8 * c);  // trunc like astype(int8)
        }
        xq32[rc * 2 + hf] = packed;
    }
    __syncthreads();

    const float scl[8] = {16.f, 32.f, 64.f, 128.f,
                          256.f, 512.f, 1024.f, -2048.f};  // 16 * bit sign

    // ---- Phase 2: 64 groups (16 rows x 4 col groups), 8 per wave ------
    for (int k = 0; k < 8; ++k) {
        const int g     = (k << 3) + waveId;   // wave-uniform
        const int trow  = g >> 2;
        const int tcolg = g & 3;
        const int oh    = oh0 + trow;
        if (oh >= OH) continue;                // uniform branch per wave

        // 12 dwords cache this lane's 48 patch bytes (4 ch per dword)
        const int baseU = (trow * IN_COLS + tcolg * 16 + mn) * 2;
        unsigned int Wd[3][2][2];
#pragma unroll
        for (int s = 0; s < 3; ++s)
#pragma unroll
            for (int t = 0; t < 2; ++t)
#pragma unroll
                for (int hf = 0; hf < 2; ++hf)
                    Wd[s][t][hf] = xq32[baseU + rc2[2 * s + t] + hf];

        float acc[8] = {0.f, 0.f, 0.f, 0.f, 0.f, 0.f, 0.f, 0.f};
#pragma unroll
        for (int n = 0; n < 8; ++n) {          // bit planes
            v8f cacc = {};
#pragma unroll
            for (int s = 0; s < 3; ++s) {      // K = 96 in 3 WMMA steps
                union { v16h v; unsigned int u[8]; } Bf;
#pragma unroll
                for (int t = 0; t < 2; ++t) {
#pragma unroll
                    for (int hf = 0; hf < 2; ++hf) {
                        const unsigned int w = Wd[s][t][hf];
                        // bytes 0&2 -> f16 pair, bytes 1&3 -> f16 pair
                        Bf.u[t * 4 + hf * 2 + 0] =
                            ((w >> n) & 0x00010001u) * 0x3C00u;
                        Bf.u[t * 4 + hf * 2 + 1] =
                            ((w >> (n + 8)) & 0x00010001u) * 0x3C00u;
                    }
                }
                cacc = __builtin_amdgcn_wmma_f32_16x16x32_f16(
                    false, A[s].v, false, Bf.v, (short)0, cacc, false, false);
            }
            // weights pre-scaled: cacc == y/16. q = clamp(rint(.)) ;
            // acc += q * 16 * sign[n]
#pragma unroll
            for (int i = 0; i < 8; ++i) {
                float q = __builtin_rintf(cacc[i]);
                q = __builtin_amdgcn_fmed3f(q, -128.f, 127.f);
                acc[i] = fmaf(q, scl[n], acc[i]);
            }
        }

        // D layout: VGPR i -> M = i + 8*laneHi, N = lane&15
        const int ow = ow0 + tcolg * 16 + mn;
        if (ow < OW) {
#pragma unroll
            for (int i = 0; i < 8; ++i) {
                const int m = i + 8 * laneHi;
                out[((b * OC + m) * OH + oh) * OW + ow] = acc[i];
            }
        }
    }
}

extern "C" void kernel_launch(void* const* d_in, const int* in_sizes, int n_in,
                              void* d_out, int out_size, void* d_ws, size_t ws_size,
                              hipStream_t stream) {
    (void)in_sizes; (void)n_in; (void)d_ws; (void)ws_size; (void)out_size;
    const float* x   = (const float*)d_in[0];   // [16,8,256,256] f32
    const float* wgt = (const float*)d_in[1];   // [16,8,3,3]     f32
    float*       out = (float*)d_out;           // [16,16,254,254] f32

    dim3 grid(4, 16, 16);   // colTiles, rowTiles, batch
    dim3 block(256);        // 8 wave32
    xbarconv_wmma_kernel<<<grid, block, 0, stream>>>(x, wgt, out);
}
